// RecurrentClassifier_51831665328868
// MI455X (gfx1250) — matrix operand order repaired and checked by reading, not compiled, not run on hardware
//
#include <hip/hip_runtime.h>
#include <hip/hip_bf16.h>

typedef __attribute__((ext_vector_type(16))) _Float16 v16h;
typedef __attribute__((ext_vector_type(8)))  _Float16 v8h;
typedef __attribute__((ext_vector_type(8)))  float    v8f;
typedef __attribute__((ext_vector_type(2)))  float    v2f;

#define B_DIM 4096
#define D_DIM 2048
#define H_DIM 1024
#define C_DIM 64
#define T_STEPS 64
#define BETA 0.9f
#define THR 1.0f

// ---------------------------------------------------------------------------
// WMMA fragment loaders (contiguous k order, same for A and B; not the ISA table order)
//   A 16x32: lane(li,hi) holds K=[k0+16hi,+16) of row li
//   B 32x16: lane(li,hi) holds K=[k0+16hi,+16) of column li (weight row li)
// ---------------------------------------------------------------------------
__device__ __forceinline__ v16h load_a_frag(const _Float16* row, int k0, int hi) {
    return *(const v16h*)(row + k0 + 16 * hi);
}
__device__ __forceinline__ v16h load_b_frag(const _Float16* row, int k0, int hi) {
    return *(const v16h*)(row + k0 + 16 * hi);   // one contiguous 32B chunk
}

// ---------------------------------------------------------------------------
// f32 -> f16 weight conversion (one-time per call; cheap)
// ---------------------------------------------------------------------------
__global__ void cvt_f32_to_f16(const float* __restrict__ src,
                               _Float16* __restrict__ dst, int n) {
    int i = blockIdx.x * blockDim.x + threadIdx.x;
    if (i < n) dst[i] = (_Float16)src[i];
}

// ---------------------------------------------------------------------------
// Kernel 0: cur1 = x @ W1^T + b1   (f32 WMMA 16x16x4, full precision, one-time)
// ---------------------------------------------------------------------------
__global__ void cur1_gemm_f32(const float* __restrict__ x,
                              const float* __restrict__ W1,
                              const float* __restrict__ b1,
                              float* __restrict__ cur1) {
    const int lane = threadIdx.x & 31;
    const int wave = threadIdx.x >> 5;
    const int flat = blockIdx.x * 8 + wave;      // [0, 16384)
    const int m0 = (flat >> 6) * 16;             // 256 M tiles
    const int n0 = (flat & 63) * 16;             // 64 N tiles
    const int li = lane & 15, hi = lane >> 4;

    const float* arow = x  + (size_t)(m0 + li) * D_DIM;
    const float* brow = W1 + (size_t)(n0 + li) * D_DIM;

    v8f acc = {};
    for (int k0 = 0; k0 < D_DIM; k0 += 4) {
        v2f a = *(const v2f*)(arow + k0 + 2 * hi);
        v2f b = *(const v2f*)(brow + k0 + 2 * hi);
        acc = __builtin_amdgcn_wmma_f32_16x16x4_f32(
            false, a, false, b, (short)0, acc, false, false);
    }

    const float bias = b1[n0 + li];
#pragma unroll
    for (int r = 0; r < 8; ++r) {
        int m = m0 + hi * 8 + r;
        cur1[(size_t)m * H_DIM + n0 + li] = acc[r] + bias;
    }
}

// ---------------------------------------------------------------------------
// Kernel 1 (per step): mem1n = beta*mem1 + cur1 + spk1@Wr1^T + br1 - r1*THR
//                      spk1n = (mem1n > THR)
// Register-blocked: each wave computes a 32(M)x64(N) macro-tile = 2x4 WMMA
// tiles -> 12 b128 loads per 8 WMMAs. Block = 8 waves covering 128x128.
// Grid: (1024/128, 4096/128) = (8, 32), 256 threads.
// ---------------------------------------------------------------------------
__global__ void layer1_step(const _Float16* __restrict__ spk_prev,
                            _Float16* __restrict__ spk_new,
                            float* __restrict__ mem1,
                            const float* __restrict__ cur1,
                            const _Float16* __restrict__ Wr1h,
                            const float* __restrict__ br1) {
    const int lane = threadIdx.x & 31;
    const int wave = threadIdx.x >> 5;           // 0..7
    const int m0 = blockIdx.y * 128 + (wave >> 1) * 32;   // 2 M-tiles of 16
    const int n0 = blockIdx.x * 128 + (wave & 1) * 64;    // 4 N-tiles of 16
    const int li = lane & 15, hi = lane >> 4;

    const _Float16* arow0 = spk_prev + (size_t)(m0 + li) * H_DIM;
    const _Float16* arow1 = arow0 + (size_t)16 * H_DIM;
    const _Float16* brow0 = Wr1h + (size_t)(n0 + li) * H_DIM;
    const _Float16* brow1 = brow0 + (size_t)16 * H_DIM;
    const _Float16* brow2 = brow0 + (size_t)32 * H_DIM;
    const _Float16* brow3 = brow0 + (size_t)48 * H_DIM;

    v8f acc[2][4] = {};
    for (int k0 = 0; k0 < H_DIM; k0 += 32) {
        __builtin_prefetch(arow0 + k0 + 128, 0, 0);
        __builtin_prefetch(brow0 + k0 + 128, 0, 0);
        v16h a0 = load_a_frag(arow0, k0, hi);
        v16h a1 = load_a_frag(arow1, k0, hi);
        v16h b0 = load_b_frag(brow0, k0, hi);
        v16h b1 = load_b_frag(brow1, k0, hi);
        v16h b2 = load_b_frag(brow2, k0, hi);
        v16h b3 = load_b_frag(brow3, k0, hi);
        acc[0][0] = __builtin_amdgcn_wmma_f32_16x16x32_f16(false, a0, false, b0, (short)0, acc[0][0], false, false);
        acc[0][1] = __builtin_amdgcn_wmma_f32_16x16x32_f16(false, a0, false, b1, (short)0, acc[0][1], false, false);
        acc[0][2] = __builtin_amdgcn_wmma_f32_16x16x32_f16(false, a0, false, b2, (short)0, acc[0][2], false, false);
        acc[0][3] = __builtin_amdgcn_wmma_f32_16x16x32_f16(false, a0, false, b3, (short)0, acc[0][3], false, false);
        acc[1][0] = __builtin_amdgcn_wmma_f32_16x16x32_f16(false, a1, false, b0, (short)0, acc[1][0], false, false);
        acc[1][1] = __builtin_amdgcn_wmma_f32_16x16x32_f16(false, a1, false, b1, (short)0, acc[1][1], false, false);
        acc[1][2] = __builtin_amdgcn_wmma_f32_16x16x32_f16(false, a1, false, b2, (short)0, acc[1][2], false, false);
        acc[1][3] = __builtin_amdgcn_wmma_f32_16x16x32_f16(false, a1, false, b3, (short)0, acc[1][3], false, false);
    }

#pragma unroll
    for (int j = 0; j < 4; ++j) {
        const int n = n0 + 16 * j + li;
        const float bias = br1[n];
#pragma unroll
        for (int i = 0; i < 2; ++i) {
#pragma unroll
            for (int r = 0; r < 8; ++r) {
                int m = m0 + 16 * i + hi * 8 + r;
                size_t idx = (size_t)m * H_DIM + n;
                float mold = mem1[idx];
                float rst  = (mold > THR) ? THR : 0.0f;   // reset uses previous mem
                float mnew = BETA * mold + cur1[idx] + acc[i][j][r] + bias - rst;
                mem1[idx] = mnew;
                spk_new[idx] = (_Float16)((mnew > THR) ? 1.0f : 0.0f);
            }
        }
    }
}

// ---------------------------------------------------------------------------
// Kernel 2 (per step): cur2 = spk1n@W2^T + b2 ; mem2n = beta*mem2 + cur2 +
//   spk2@Wr2^T + br2 - r2*THR ; spk2n = (mem2n > THR). Writes output slices.
// N=64 is small: keep 16x16 per wave for parallelism (1024 waves).
// ---------------------------------------------------------------------------
__global__ void layer2_step(const _Float16* __restrict__ spk1,
                            const _Float16* __restrict__ spk2_prev,
                            _Float16* __restrict__ spk2_new,
                            float* __restrict__ mem2,
                            const _Float16* __restrict__ W2h,
                            const float* __restrict__ b2,
                            const _Float16* __restrict__ Wr2h,
                            const float* __restrict__ br2,
                            float* __restrict__ spk_out,
                            float* __restrict__ mem_out) {
    const int lane = threadIdx.x & 31;
    const int wave = threadIdx.x >> 5;
    const int flat = blockIdx.x * 8 + wave;      // [0, 1024)
    const int m0 = (flat >> 2) * 16;             // 256 M tiles
    const int n0 = (flat & 3) * 16;              // 4 N tiles
    const int li = lane & 15, hi = lane >> 4;

    v8f acc = {};
    {   // feed-forward: K = 1024 over spk1 / W2
        const _Float16* arow = spk1 + (size_t)(m0 + li) * H_DIM;
        const _Float16* brow = W2h  + (size_t)(n0 + li) * H_DIM;
        for (int k0 = 0; k0 < H_DIM; k0 += 32) {
            v16h a = load_a_frag(arow, k0, hi);
            v16h b = load_b_frag(brow, k0, hi);
            acc = __builtin_amdgcn_wmma_f32_16x16x32_f16(
                false, a, false, b, (short)0, acc, false, false);
        }
    }
    {   // recurrent: K = 64 over spk2_prev / Wr2
        const _Float16* arow = spk2_prev + (size_t)(m0 + li) * C_DIM;
        const _Float16* brow = Wr2h      + (size_t)(n0 + li) * C_DIM;
#pragma unroll
        for (int k0 = 0; k0 < C_DIM; k0 += 32) {
            v16h a = load_a_frag(arow, k0, hi);
            v16h b = load_b_frag(brow, k0, hi);
            acc = __builtin_amdgcn_wmma_f32_16x16x32_f16(
                false, a, false, b, (short)0, acc, false, false);
        }
    }

    const float bias = b2[n0 + li] + br2[n0 + li];
#pragma unroll
    for (int r = 0; r < 8; ++r) {
        int m = m0 + hi * 8 + r;
        size_t idx = (size_t)m * C_DIM + n0 + li;
        float mold = mem2[idx];
        float rst  = (mold > THR) ? THR : 0.0f;
        float mnew = BETA * mold + acc[r] + bias - rst;
        float spk  = (mnew > THR) ? 1.0f : 0.0f;
        mem2[idx] = mnew;
        spk2_new[idx] = (_Float16)spk;
        spk_out[idx] = spk;
        mem_out[idx] = mnew;
    }
}

// ---------------------------------------------------------------------------
extern "C" void kernel_launch(void* const* d_in, const int* in_sizes, int n_in,
                              void* d_out, int out_size, void* d_ws, size_t ws_size,
                              hipStream_t stream) {
    const float* x   = (const float*)d_in[0];
    const float* W1  = (const float*)d_in[1];
    const float* b1  = (const float*)d_in[2];
    const float* Wr1 = (const float*)d_in[3];
    const float* br1 = (const float*)d_in[4];
    const float* W2  = (const float*)d_in[5];
    const float* b2  = (const float*)d_in[6];
    const float* Wr2 = (const float*)d_in[7];
    const float* br2 = (const float*)d_in[8];
    float* out = (float*)d_out;

    // workspace carve-up (all chunks 256B-aligned by construction)
    char* ws = (char*)d_ws;
    size_t off = 0;
    auto carve = [&](size_t bytes) -> void* {
        void* p = ws + off;
        off += (bytes + 255) & ~(size_t)255;
        return p;
    };
    float*    cur1   = (float*)   carve((size_t)B_DIM * H_DIM * 4);
    float*    mem1   = (float*)   carve((size_t)B_DIM * H_DIM * 4);
    float*    mem2   = (float*)   carve((size_t)B_DIM * C_DIM * 4);
    _Float16* spk1A  = (_Float16*)carve((size_t)B_DIM * H_DIM * 2);
    _Float16* spk1B  = (_Float16*)carve((size_t)B_DIM * H_DIM * 2);
    _Float16* spk2A  = (_Float16*)carve((size_t)B_DIM * C_DIM * 2);
    _Float16* spk2B  = (_Float16*)carve((size_t)B_DIM * C_DIM * 2);
    _Float16* Wr1h   = (_Float16*)carve((size_t)H_DIM * H_DIM * 2);
    _Float16* W2h    = (_Float16*)carve((size_t)C_DIM * H_DIM * 2);
    _Float16* Wr2h   = (_Float16*)carve((size_t)C_DIM * C_DIM * 2);

    // zero the recurrent state (deterministic per call)
    hipMemsetAsync(mem1,  0, (size_t)B_DIM * H_DIM * 4, stream);
    hipMemsetAsync(mem2,  0, (size_t)B_DIM * C_DIM * 4, stream);
    hipMemsetAsync(spk1A, 0, (size_t)B_DIM * H_DIM * 2, stream);
    hipMemsetAsync(spk2A, 0, (size_t)B_DIM * C_DIM * 2, stream);

    // weight conversions
    cvt_f32_to_f16<<<(H_DIM * H_DIM + 255) / 256, 256, 0, stream>>>(Wr1, Wr1h, H_DIM * H_DIM);
    cvt_f32_to_f16<<<(C_DIM * H_DIM + 255) / 256, 256, 0, stream>>>(W2,  W2h,  C_DIM * H_DIM);
    cvt_f32_to_f16<<<(C_DIM * C_DIM + 255) / 256, 256, 0, stream>>>(Wr2, Wr2h, C_DIM * C_DIM);

    // one-time input GEMM (f32 WMMA)
    cur1_gemm_f32<<<2048, 256, 0, stream>>>(x, W1, b1, cur1);

    // time-step loop: 2 fused kernels per step
    for (int t = 0; t < T_STEPS; ++t) {
        _Float16* s1p = (t & 1) ? spk1B : spk1A;
        _Float16* s1n = (t & 1) ? spk1A : spk1B;
        _Float16* s2p = (t & 1) ? spk2B : spk2A;
        _Float16* s2n = (t & 1) ? spk2A : spk2B;

        layer1_step<<<dim3(8, 32), 256, 0, stream>>>(s1p, s1n, mem1, cur1, Wr1h, br1);

        float* spk_slice = out + (size_t)t * B_DIM * C_DIM;
        float* mem_slice = out + (size_t)T_STEPS * B_DIM * C_DIM + (size_t)t * B_DIM * C_DIM;
        layer2_step<<<128, 256, 0, stream>>>(s1n, s2p, s2n, mem2, W2h, b2, Wr2h, br2,
                                             spk_slice, mem_slice);
    }
}
